// GNNDecoder_65970697666774
// MI455X (gfx1250) — matrix operand order repaired and checked by reading, not compiled, not run on hardware
//
#include <hip/hip_runtime.h>
#include <hip/hip_bf16.h>

typedef __attribute__((ext_vector_type(16))) __bf16 v16bf;
typedef __attribute__((ext_vector_type(8)))  float  v8f;

#define N_GRAPHS 100
#define EMB_D    128
#define G1_D     512
#define G2_D     64000
#define N_NODESC 100000
#define N_EDGESC 1600000
#define DD       64
#define ECH      16
#define NM1_D    128

// ---------------------------------------------------------------------------
// WMMA helpers (CDNA5 wave32, 16x16x32 bf16 -> f32 accumulate)
// ---------------------------------------------------------------------------
__device__ __forceinline__ v8f wmma_bf16(v16bf a, v16bf b, v8f c) {
  return __builtin_amdgcn_wmma_f32_16x16x32_bf16(false, a, false, b, (short)0, c,
                                                 false, false);
}

// K offset (0..31) inside a 32-wide chunk for 16-bit A fragment element e.
__device__ __forceinline__ int a_k(int lane, int e) {
  int v = e >> 1, h = e & 1;
  return ((v >> 2) << 4) | (((lane >> 4) & 1) << 3) | ((v & 3) << 1) | h;
}
// K offset (0..31) inside a 32-wide chunk for 16-bit B fragment element e.
__device__ __forceinline__ int b_k(int lane, int e) {
  int v = e >> 1, h = e & 1;
  return ((v >> 2) << 4) | (((lane >> 4) & 1) << 3) | ((v & 3) << 1) | h;
}

__device__ __forceinline__ void atomic_add_f32(float* p, float v) {
  __hip_atomic_fetch_add(p, v, __ATOMIC_RELAXED, __HIP_MEMORY_SCOPE_AGENT);
}

// ---------------------------------------------------------------------------
// inner_dec layer 1: h1 = relu(emb @ Wg1 + bg1)   [100,128]@[128,512]
// ---------------------------------------------------------------------------
__global__ void mlp1_kernel(const float* __restrict__ emb,
                            const float* __restrict__ Wg1,
                            const float* __restrict__ bg1,
                            float* __restrict__ h1) {
  int wid  = (blockIdx.x * blockDim.x + threadIdx.x) >> 5;
  int lane = threadIdx.x & 31;
  const int NT = G1_D / 16;             // 32
  const int MT = (N_GRAPHS + 15) / 16;  // 7
  int mt = wid / NT, nt = wid % NT;
  if (mt >= MT) return;
  int n15  = lane & 15;
  int mrow = min(mt * 16 + n15, N_GRAPHS - 1);
  int ncol = nt * 16 + n15;
  v8f c = {};
#pragma unroll
  for (int kc = 0; kc < EMB_D / 32; ++kc) {
    v16bf a, b;
#pragma unroll
    for (int e = 0; e < 16; ++e) {
      a[e] = (__bf16)emb[mrow * EMB_D + kc * 32 + a_k(lane, e)];
      b[e] = (__bf16)Wg1[(kc * 32 + b_k(lane, e)) * G1_D + ncol];
    }
    c = wmma_bf16(a, b, c);
  }
#pragma unroll
  for (int v = 0; v < 8; ++v) {
    int m = mt * 16 + ((lane >> 4) << 3) + v;
    float val = fmaxf(c[v] + bg1[ncol], 0.0f);
    if (m < N_GRAPHS) h1[m * G1_D + ncol] = val;
  }
}

// ---------------------------------------------------------------------------
// inner_dec layer 2: x0 = h1 @ Wg2 + bg2  [100,512]@[512,64000].
// One wave per 16-column tile; loops all 7 M-tiles with the B fragment in
// registers so Wg2 (131 MB) is streamed exactly once.
// ---------------------------------------------------------------------------
__global__ void mlp2_kernel(const float* __restrict__ h1,
                            const float* __restrict__ Wg2,
                            const float* __restrict__ bg2,
                            float* __restrict__ x0) {
  int wid  = (blockIdx.x * blockDim.x + threadIdx.x) >> 5;
  int lane = threadIdx.x & 31;
  const int NT = G2_D / 16;  // 4000
  if (wid >= NT) return;
  int n15  = lane & 15;
  int ncol = wid * 16 + n15;
  v8f c[7] = {};
  for (int kc = 0; kc < G1_D / 32; ++kc) {  // 16 chunks
    v16bf b;
#pragma unroll
    for (int e = 0; e < 16; ++e)
      b[e] = (__bf16)Wg2[(kc * 32 + b_k(lane, e)) * G2_D + ncol];
#pragma unroll
    for (int mt = 0; mt < 7; ++mt) {
      int mrow = min(mt * 16 + n15, N_GRAPHS - 1);
      v16bf a;
#pragma unroll
      for (int e = 0; e < 16; ++e)
        a[e] = (__bf16)h1[mrow * G1_D + kc * 32 + a_k(lane, e)];
      c[mt] = wmma_bf16(a, b, c[mt]);
    }
  }
  float bias = bg2[ncol];
#pragma unroll
  for (int mt = 0; mt < 7; ++mt)
#pragma unroll
    for (int v = 0; v < 8; ++v) {
      int m = mt * 16 + ((lane >> 4) << 3) + v;
      if (m < N_GRAPHS) x0[m * G2_D + ncol] = c[mt][v] + bias;
    }
}

// ---------------------------------------------------------------------------
// GNN message + scatter-add: m = relu([x[src]|edge_attr] @ mW + mb);
// agg[dst] += m.  K=80 padded to 96.  Grid-stride: ~8 tiles/wave so the 12
// register-resident B fragments are amortized.
// ---------------------------------------------------------------------------
__global__ void message_kernel(const float* __restrict__ x,
                               const float* __restrict__ ea,
                               const long long* __restrict__ src,
                               const long long* __restrict__ dst,
                               const float* __restrict__ mW,
                               const float* __restrict__ mb,
                               float* __restrict__ agg) {
  int gwid = (blockIdx.x * blockDim.x + threadIdx.x) >> 5;
  int nw   = (gridDim.x * blockDim.x) >> 5;
  int lane = threadIdx.x & 31;
  const int TILES = N_EDGESC / 16;  // 100000
  int n15 = lane & 15;

  v16bf Bf[3][4];
#pragma unroll
  for (int kc = 0; kc < 3; ++kc)
#pragma unroll
    for (int nt = 0; nt < 4; ++nt)
#pragma unroll
      for (int e = 0; e < 16; ++e) {
        int k = kc * 32 + b_k(lane, e);
        Bf[kc][nt][e] = (k < DD + ECH) ? (__bf16)mW[k * DD + nt * 16 + n15]
                                       : (__bf16)0.0f;
      }
  float bias[4];
#pragma unroll
  for (int nt = 0; nt < 4; ++nt) bias[nt] = mb[nt * 16 + n15];

  for (int tile = gwid; tile < TILES; tile += nw) {
    int ebase = tile * 16;
    __builtin_prefetch(&ea[(size_t)(ebase + 16 * nw) * ECH], 0, 1);
    long long s = src[ebase + n15];

    v16bf A0, A1, A2;
#pragma unroll
    for (int e = 0; e < 16; ++e) {
      int k = a_k(lane, e);
      A0[e] = (__bf16)x[s * DD + k];
      A1[e] = (__bf16)x[s * DD + 32 + k];
      A2[e] = (k < ECH) ? (__bf16)ea[(size_t)(ebase + n15) * ECH + k]
                        : (__bf16)0.0f;
    }

    v8f cacc[4] = {};
#pragma unroll
    for (int nt = 0; nt < 4; ++nt) {
      cacc[nt] = wmma_bf16(A0, Bf[0][nt], cacc[nt]);
      cacc[nt] = wmma_bf16(A1, Bf[1][nt], cacc[nt]);
      cacc[nt] = wmma_bf16(A2, Bf[2][nt], cacc[nt]);
    }

#pragma unroll
    for (int nt = 0; nt < 4; ++nt) {
      int n = nt * 16 + n15;
#pragma unroll
      for (int v = 0; v < 8; ++v) {
        int m = ((lane >> 4) << 3) + v;
        long long dnode = dst[ebase + m];
        float val = fmaxf(cacc[nt][v] + bias[nt], 0.0f);
        atomic_add_f32(&agg[dnode * DD + n], val);
      }
    }
  }
}

// ---------------------------------------------------------------------------
// GNN update: x_out = relu([x | agg] @ uW + ub), K=128, 16 nodes/tile.
// Per-tile A data (two contiguous 4 KB row blocks) is staged into LDS with
// CDNA5 async loads (global_load_async_to_lds_b128 + s_wait_asynccnt), then
// converted to bf16 fragments.  x_out may alias agg (reads precede stores
// within a wave; tiles partition rows).
// ---------------------------------------------------------------------------
__global__ void update_kernel(const float* __restrict__ xin,
                              const float* __restrict__ agg,
                              const float* __restrict__ uW,
                              const float* __restrict__ ub,
                              float* __restrict__ xout) {
  __shared__ float sx[4][16][DD];  // per-wave 4 KB
  __shared__ float sa[4][16][DD];  // per-wave 4 KB
  int wib  = threadIdx.x >> 5;
  int lane = threadIdx.x & 31;
  int gwid = (blockIdx.x * blockDim.x + threadIdx.x) >> 5;
  int nw   = (gridDim.x * blockDim.x) >> 5;
  const int TILES = N_NODESC / 16;  // 6250
  int n15 = lane & 15;

  v16bf Bf[4][4];
#pragma unroll
  for (int kc = 0; kc < 4; ++kc)
#pragma unroll
    for (int nt = 0; nt < 4; ++nt)
#pragma unroll
      for (int e = 0; e < 16; ++e)
        Bf[kc][nt][e] = (__bf16)uW[(kc * 32 + b_k(lane, e)) * DD + nt * 16 + n15];

  unsigned lx = (unsigned)(size_t)&sx[wib][0][0];
  unsigned la = (unsigned)(size_t)&sa[wib][0][0];

  for (int tile = gwid; tile < TILES; tile += nw) {
    const float* gx = xin + (size_t)tile * 16 * DD;
    const float* ga = agg + (size_t)tile * 16 * DD;
    // async-stage 2 x 4KB into this wave's LDS region (16 B per lane/issue)
#pragma unroll
    for (int i = 0; i < 8; ++i) {
      unsigned off = (unsigned)(i * 32 + lane) * 16u;
      unsigned long long gax = (unsigned long long)(size_t)gx + off;
      unsigned long long gaa = (unsigned long long)(size_t)ga + off;
      asm volatile("global_load_async_to_lds_b128 %0, %1, off"
                   :: "v"(lx + off), "v"(gax) : "memory");
      asm volatile("global_load_async_to_lds_b128 %0, %1, off"
                   :: "v"(la + off), "v"(gaa) : "memory");
    }
    asm volatile("s_wait_asynccnt 0x0" ::: "memory");

    v16bf A[4];
#pragma unroll
    for (int e = 0; e < 16; ++e) {
      int k = a_k(lane, e);
      A[0][e] = (__bf16)sx[wib][n15][k];
      A[1][e] = (__bf16)sx[wib][n15][32 + k];
      A[2][e] = (__bf16)sa[wib][n15][k];
      A[3][e] = (__bf16)sa[wib][n15][32 + k];
    }

    v8f cacc[4] = {};
#pragma unroll
    for (int kc = 0; kc < 4; ++kc) {
      cacc[0] = wmma_bf16(A[kc], Bf[kc][0], cacc[0]);
      cacc[1] = wmma_bf16(A[kc], Bf[kc][1], cacc[1]);
      cacc[2] = wmma_bf16(A[kc], Bf[kc][2], cacc[2]);
      cacc[3] = wmma_bf16(A[kc], Bf[kc][3], cacc[3]);
    }

#pragma unroll
    for (int nt = 0; nt < 4; ++nt) {
      int n = nt * 16 + n15;
      float bias = ub[n];
#pragma unroll
      for (int v = 0; v < 8; ++v) {
        int m = tile * 16 + ((lane >> 4) << 3) + v;
        xout[m * DD + n] = fmaxf(cacc[nt][v] + bias, 0.0f);
      }
    }
  }
}

// ---------------------------------------------------------------------------
// node_mlp: out = relu(x @ nW1 + nb1) @ nW2 + nb2.
// 8 block-uniform iterations of 4 tiles; nW1 fragments hoisted once per wave.
// ---------------------------------------------------------------------------
__global__ void node_mlp_kernel(const float* __restrict__ x,
                                const float* __restrict__ nW1,
                                const float* __restrict__ nb1,
                                const float* __restrict__ nW2,
                                const float* __restrict__ nb2,
                                float* __restrict__ out) {
  __shared__ float lds[4][16][NM1_D];  // 32 KB
  int wib  = threadIdx.x >> 5;
  int lane = threadIdx.x & 31;
  const int TILES = N_NODESC / 16;  // 6250
  int n15 = lane & 15;

  v16bf Bf[2][8];
#pragma unroll
  for (int kc = 0; kc < 2; ++kc)
#pragma unroll
    for (int nt = 0; nt < 8; ++nt)
#pragma unroll
      for (int e = 0; e < 16; ++e)
        Bf[kc][nt][e] = (__bf16)nW1[(kc * 32 + b_k(lane, e)) * NM1_D + nt * 16 + n15];

  for (int it = 0; it < 8; ++it) {  // uniform trip count across block
    int tile = blockIdx.x * 32 + it * 4 + wib;
    bool active = tile < TILES;
    if (active) {
      int row = tile * 16 + n15;
      v16bf A0, A1;
#pragma unroll
      for (int e = 0; e < 16; ++e) {
        int k = a_k(lane, e);
        A0[e] = (__bf16)x[row * DD + k];
        A1[e] = (__bf16)x[row * DD + 32 + k];
      }
      v8f c[8] = {};
#pragma unroll
      for (int nt = 0; nt < 8; ++nt) {
        c[nt] = wmma_bf16(A0, Bf[0][nt], c[nt]);
        c[nt] = wmma_bf16(A1, Bf[1][nt], c[nt]);
      }
#pragma unroll
      for (int nt = 0; nt < 8; ++nt) {
        int n = nt * 16 + n15;
        float bias = nb1[n];
#pragma unroll
        for (int v = 0; v < 8; ++v)
          lds[wib][((lane >> 4) << 3) + v][n] = fmaxf(c[nt][v] + bias, 0.0f);
      }
    }
    __syncthreads();
    if (active && lane < 16) {
      int m = lane;
      int node = tile * 16 + m;
      float a0 = nb2[0], a1 = nb2[1], a2 = nb2[2];
#pragma unroll 8
      for (int cc = 0; cc < NM1_D; ++cc) {
        float hv = lds[wib][m][cc];
        a0 += hv * nW2[cc * 3 + 0];
        a1 += hv * nW2[cc * 3 + 1];
        a2 += hv * nW2[cc * 3 + 2];
      }
      out[node * 3 + 0] = a0;
      out[node * 3 + 1] = a1;
      out[node * 3 + 2] = a2;
    }
    __syncthreads();
  }
}

// ---------------------------------------------------------------------------
extern "C" void kernel_launch(void* const* d_in, const int* in_sizes, int n_in,
                              void* d_out, int out_size, void* d_ws, size_t ws_size,
                              hipStream_t stream) {
  const float*     emb  = (const float*)d_in[0];
  const long long* eidx = (const long long*)d_in[1];  // int64 [2, E]
  const float*     ea   = (const float*)d_in[2];
  const float*     Wg1  = (const float*)d_in[3];
  const float*     bg1  = (const float*)d_in[4];
  const float*     Wg2  = (const float*)d_in[5];
  const float*     bg2  = (const float*)d_in[6];
  const float*     mW0  = (const float*)d_in[7];
  const float*     mb0  = (const float*)d_in[8];
  const float*     uW0  = (const float*)d_in[9];
  const float*     ub0  = (const float*)d_in[10];
  const float*     mW1  = (const float*)d_in[11];
  const float*     mb1  = (const float*)d_in[12];
  const float*     uW1  = (const float*)d_in[13];
  const float*     ub1  = (const float*)d_in[14];
  const float*     nW1  = (const float*)d_in[15];
  const float*     nb1  = (const float*)d_in[16];
  const float*     nW2  = (const float*)d_in[17];
  const float*     nb2  = (const float*)d_in[18];
  float*           out  = (float*)d_out;

  float* buf0 = (float*)d_ws;                  // [N_NODES * 64]
  float* buf1 = buf0 + (size_t)N_NODESC * DD;  // [N_NODES * 64]
  float* h1   = buf1 + (size_t)N_NODESC * DD;  // [100 * 512]

  const long long* src = eidx;
  const long long* dst = eidx + N_EDGESC;

  // inner_dec
  mlp1_kernel<<<28, 256, 0, stream>>>(emb, Wg1, bg1, h1);     // 224 waves
  mlp2_kernel<<<500, 256, 0, stream>>>(h1, Wg2, bg2, buf0);   // 4000 waves

  // GNN layer 0: x=buf0, agg=buf1 -> x=buf1
  hipMemsetAsync(buf1, 0, sizeof(float) * (size_t)N_NODESC * DD, stream);
  message_kernel<<<1563, 256, 0, stream>>>(buf0, ea, src, dst, mW0, mb0, buf1);
  update_kernel<<<196, 128, 0, stream>>>(buf0, buf1, uW0, ub0, buf1);

  // GNN layer 1: x=buf1, agg=buf0 -> x=buf0
  hipMemsetAsync(buf0, 0, sizeof(float) * (size_t)N_NODESC * DD, stream);
  message_kernel<<<1563, 256, 0, stream>>>(buf1, ea, src, dst, mW1, mb1, buf0);
  update_kernel<<<196, 128, 0, stream>>>(buf1, buf0, uW1, ub1, buf0);

  // node reconstruction MLP
  node_mlp_kernel<<<196, 128, 0, stream>>>(buf0, nW1, nb1, nW2, nb2, out);
}
